// GroupedQueryAttention_58394375357343
// MI455X (gfx1250) — matrix operand order repaired and checked by reading, not compiled, not run on hardware
//
#include <hip/hip_runtime.h>

// ---------------------------------------------------------------------------
// CDNA5 (gfx1250, wave32) GQA attention block.
// BF16 WMMA (v_wmma_f32_16x16x32_bf16) for all matmuls, f32 accumulate.
// ---------------------------------------------------------------------------

typedef __attribute__((ext_vector_type(16))) __bf16 bf16x16;
typedef __attribute__((ext_vector_type(8)))  __bf16 bf16x8;
typedef __attribute__((ext_vector_type(8)))  float  floatx8;

#define BATCH 2
#define SEQ   2048
#define DIM   1024
#define HEADS 16
#define DHEAD 64
#define KVHEADS 4
#define KVDIM 256
#define ROWS  (BATCH * SEQ)       // 4096
#define ATTN_SCALE 0.125f         // 1/sqrt(64)

__device__ __forceinline__ unsigned short f2bf(float f) {
    unsigned u = __float_as_uint(f);
    unsigned r = u + 0x7FFFu + ((u >> 16) & 1u);   // round-to-nearest-even
    return (unsigned short)(r >> 16);
}

__device__ __forceinline__ floatx8 wmma_bf16(bf16x16 a, bf16x16 b, floatx8 c) {
    // (neg_a, A, neg_b, B, c_mod, C, reuse_a, reuse_b)
    return __builtin_amdgcn_wmma_f32_16x16x32_bf16(false, a, false, b,
                                                   (short)0, c, false, false);
}

union FragU { bf16x8 h2[2]; bf16x16 v; };

// Load a 16x32 A-fragment (lane m = lane%16; lanes<16 hold
// K 0..15, lanes>=16 hold K 16..31).
__device__ __forceinline__ bf16x16 load_frag_a(const unsigned short* rowBase, int half) {
    FragU u;
    u.h2[0] = *(const bf16x8*)(rowBase + half * 16);
    u.h2[1] = *(const bf16x8*)(rowBase + half * 16 + 8);
    return u.v;
}

// ---------------------------------------------------------------------------
// 1) Fused fp32 -> bf16 cast + transpose for weights: src[K][N] -> dst[N][K].
//    32x32 tiles through LDS (33-wide pad to dodge bank conflicts).
// ---------------------------------------------------------------------------
__global__ __launch_bounds__(256)
void castT_bf16_kernel(const float* __restrict__ src, unsigned short* __restrict__ dst,
                       int K, int N) {
    __shared__ float tile[32][33];
    const int kt = blockIdx.x * 32, nt = blockIdx.y * 32;
    const int tx = threadIdx.x & 31, ty = threadIdx.x >> 5;   // 32 x 8
#pragma unroll
    for (int i = 0; i < 4; ++i)
        tile[ty + 8 * i][tx] = src[(unsigned)(kt + ty + 8 * i) * N + nt + tx];
    __syncthreads();
#pragma unroll
    for (int i = 0; i < 4; ++i)
        dst[(unsigned)(nt + ty + 8 * i) * K + kt + tx] = f2bf(tile[tx][ty + 8 * i]);
}

// ---------------------------------------------------------------------------
// 2) Fused LayerNorm + bf16 cast. One block per row (1024 cols, 256 threads).
// ---------------------------------------------------------------------------
__global__ __launch_bounds__(256)
void ln_bf16_kernel(const float* __restrict__ x, const float* __restrict__ gamma,
                    const float* __restrict__ beta, unsigned short* __restrict__ out) {
    __shared__ float red[256];
    const int row = blockIdx.x;
    const int t = threadIdx.x;
    const float* xr = x + (size_t)row * DIM;

    float v[4];
    float s = 0.f;
#pragma unroll
    for (int i = 0; i < 4; ++i) { v[i] = xr[t + 256 * i]; s += v[i]; }
    red[t] = s; __syncthreads();
    for (int o = 128; o > 0; o >>= 1) { if (t < o) red[t] += red[t + o]; __syncthreads(); }
    const float mu = red[0] * (1.f / DIM);
    __syncthreads();

    float s2 = 0.f;
#pragma unroll
    for (int i = 0; i < 4; ++i) { float d = v[i] - mu; s2 += d * d; }
    red[t] = s2; __syncthreads();
    for (int o = 128; o > 0; o >>= 1) { if (t < o) red[t] += red[t + o]; __syncthreads(); }
    const float rstd = rsqrtf(red[0] * (1.f / DIM) + 1e-5f);

#pragma unroll
    for (int i = 0; i < 4; ++i) {
        int col = t + 256 * i;
        float y = (v[i] - mu) * rstd * gamma[col] + beta[col];
        out[(unsigned)row * DIM + col] = f2bf(y);
    }
}

// ---------------------------------------------------------------------------
// 3) BF16 GEMM: C[M,N] = A[M,K] @ W[K,N], with W supplied transposed
//    (Bt = W^T, [N][K] bf16) so both LDS fills are b128-vectorized.
//    Block tile 128x64, K-step 32; 8 waves (4x2), wave tile 32x32 (2x2 WMMA).
//    Compile-time N, K and output mode -> branch-free shift addressing.
// ---------------------------------------------------------------------------
template <int N, int K, bool F32OUT>
__global__ __launch_bounds__(256)
void gemm_bf16_kernel(const unsigned short* __restrict__ A,
                      const unsigned short* __restrict__ Bt,
                      unsigned short* __restrict__ Cbf,
                      float* __restrict__ Cf,
                      const float* __restrict__ bias) {
    __shared__ __align__(64) unsigned short sA[128 * 32];   // [m][k]
    __shared__ __align__(64) unsigned short sB[64 * 32];    // [n][k]

    const int t = threadIdx.x;
    const int lane = t & 31, w = t >> 5;
    const int wm = w >> 1, wn = w & 1;
    const int bm = blockIdx.x * 128, bn = blockIdx.y * 64;
    const int l16 = lane & 15, half = lane >> 4;

    // Per-thread A-load mapping: 512 16B chunks, 2 per thread
    const int arow0 = t >> 2, ako = (t & 3) << 3;            // chunk t
    const int arow1 = (t + 256) >> 2;                        // chunk t+256 (same ako)
    // Per-thread B-load mapping: 256 16B chunks, 1 per thread
    const int brow = t >> 2, bko = (t & 3) << 3;

    floatx8 acc[2][2] = {};

    for (int k0 = 0; k0 < K; k0 += 32) {
        *(uint4*)&sA[arow0 * 32 + ako] =
            *(const uint4*)&A[(unsigned)(bm + arow0) * K + k0 + ako];
        *(uint4*)&sA[arow1 * 32 + ako] =
            *(const uint4*)&A[(unsigned)(bm + arow1) * K + k0 + ako];
        *(uint4*)&sB[brow * 32 + bko] =
            *(const uint4*)&Bt[(unsigned)(bn + brow) * K + k0 + bko];
        if (k0 + 32 < K) {   // speculative prefetch of next A tile
            __builtin_prefetch(&A[(unsigned)(bm + arow0) * K + k0 + 32 + ako], 0, 0);
            __builtin_prefetch(&A[(unsigned)(bm + arow1) * K + k0 + 32 + ako], 0, 0);
        }
        __syncthreads();

        bf16x16 af[2], bg[2];
#pragma unroll
        for (int i = 0; i < 2; ++i)
            af[i] = load_frag_a(&sA[(wm * 32 + i * 16 + l16) * 32], half);
#pragma unroll
        for (int j = 0; j < 2; ++j)
            bg[j] = *(const bf16x16*)&sB[(wn * 32 + j * 16 + l16) * 32 + half * 16];
#pragma unroll
        for (int i = 0; i < 2; ++i)
#pragma unroll
            for (int j = 0; j < 2; ++j)
                acc[i][j] = wmma_bf16(af[i], bg[j], acc[i][j]);
        __syncthreads();
    }

    // Epilogue. C layout: VGPR r -> row m = r + 8*(lane/16), col = lane%16.
#pragma unroll
    for (int i = 0; i < 2; ++i) {
        const int row0 = bm + wm * 32 + i * 16 + half * 8;
#pragma unroll
        for (int j = 0; j < 2; ++j) {
            const int col = bn + wn * 32 + j * 16 + l16;
            float bz = 0.f;
            if constexpr (F32OUT) bz = bias[col];
#pragma unroll
            for (int r = 0; r < 8; ++r) {
                const unsigned idx = (unsigned)(row0 + r) * N + col;
                const float vv = acc[i][j][r];
                if constexpr (F32OUT) Cf[idx] = vv + bz;
                else                  Cbf[idx] = f2bf(vv);
            }
        }
    }
}

// ---------------------------------------------------------------------------
// 4) Flash attention (no mask). Block = 4 waves = the 4 query heads sharing
//    one KV head; grid over (b, kv-head, 16-row q tile). Key tiles of 32.
// ---------------------------------------------------------------------------
__global__ __launch_bounds__(128)
void attn_kernel(const unsigned short* __restrict__ Q,
                 const unsigned short* __restrict__ Kb,
                 const unsigned short* __restrict__ Vb,
                 unsigned short* __restrict__ O) {
    __shared__ __align__(64) unsigned short sK[32 * 64];       // [key][d]
    __shared__ __align__(64) unsigned short sVt[64 * 32];      // [d][key]
    __shared__ __align__(64) unsigned short sQ[4][16 * 64];    // per-wave [m][d]
    __shared__ __align__(64) unsigned short sP[4][16 * 32];    // per-wave [m][key]

    const int t = threadIdx.x, lane = t & 31, w = t >> 5;
    const int l16 = lane & 15, half = lane >> 4;
    const int bid = blockIdx.x;
    const int qt = bid & 127;            // q tile (16 rows each)
    const int gh = (bid >> 7) & 3;       // kv head
    const int b  = bid >> 9;
    const int h  = gh * 4 + w;           // this wave's query head (g-major)

    // Load this wave's 16x64 Q tile
#pragma unroll
    for (int j = 0; j < 4; ++j) {
        int c = lane + 32 * j;           // 128 8-elem chunks per wave
        int m = c >> 3, doff = (c & 7) << 3;
        *(uint4*)&sQ[w][m * 64 + doff] =
            *(const uint4*)&Q[(unsigned)(b * SEQ + qt * 16 + m) * DIM + h * DHEAD + doff];
    }
    __syncthreads();

    bf16x16 aq[2];
#pragma unroll
    for (int i = 0; i < 2; ++i)          // d 0..31 and 32..63
        aq[i] = load_frag_a(&sQ[w][l16 * 64 + i * 32], half);

    floatx8 o4[4] = {};
    float Mrow[8], Lrow[8];
#pragma unroll
    for (int r = 0; r < 8; ++r) { Mrow[r] = -1e30f; Lrow[r] = 0.f; }

    // K-tile load mapping (2 chunks/thread) and V-transpose mapping
    const int kkey0 = t >> 3,          kdo0 = (t & 7) << 3;
    const int kkey1 = (t + 128) >> 3;  // same kdo0
    const int vkey = t >> 2, vd0 = (t & 3) * 16;

    for (int kt = 0; kt < SEQ; kt += 32) {
        __syncthreads();
        // K tile [32 key][64 d], shared by all 4 waves
        *(uint4*)&sK[kkey0 * 64 + kdo0] =
            *(const uint4*)&Kb[(unsigned)(b * SEQ + kt + kkey0) * KVDIM + gh * DHEAD + kdo0];
        *(uint4*)&sK[kkey1 * 64 + kdo0] =
            *(const uint4*)&Kb[(unsigned)(b * SEQ + kt + kkey1) * KVDIM + gh * DHEAD + kdo0];
        // V tile transposed: sVt[d][key]
        {
            const unsigned short* vsrc =
                &Vb[(unsigned)(b * SEQ + kt + vkey) * KVDIM + gh * DHEAD + vd0];
            union { uint4 q[2]; unsigned short s[16]; } tv;
            tv.q[0] = *(const uint4*)&vsrc[0];
            tv.q[1] = *(const uint4*)&vsrc[8];
#pragma unroll
            for (int d = 0; d < 16; ++d) sVt[(vd0 + d) * 32 + vkey] = tv.s[d];
        }
        if (kt + 32 < SEQ) {   // speculative prefetch of next K/V tiles
            __builtin_prefetch(
                &Kb[(unsigned)(b * SEQ + kt + 32 + kkey0) * KVDIM + gh * DHEAD + kdo0], 0, 0);
            __builtin_prefetch(
                &Vb[(unsigned)(b * SEQ + kt + 32 + vkey) * KVDIM + gh * DHEAD + vd0], 0, 0);
        }
        __syncthreads();

        // Scores: 16q x 32key = two 16x16 C tiles, K-dim = d (two 32-chunks)
        floatx8 sc[2] = {};
#pragma unroll
        for (int j = 0; j < 2; ++j)
#pragma unroll
            for (int i = 0; i < 2; ++i) {
                bf16x16 bk = *(const bf16x16*)&sK[(j * 16 + l16) * 64 + i * 32 + half * 16];
                sc[j] = wmma_bf16(aq[i], bk, sc[j]);
            }

        // Online softmax (rows live in half-waves; xor<16 stays in half)
        float p0[8], p1[8], alpha[8];
#pragma unroll
        for (int r = 0; r < 8; ++r) {
            float s0 = sc[0][r] * ATTN_SCALE, s1 = sc[1][r] * ATTN_SCALE;
            float mx = fmaxf(s0, s1);
#pragma unroll
            for (int mk = 1; mk < 16; mk <<= 1) mx = fmaxf(mx, __shfl_xor(mx, mk, 32));
            float mnew = fmaxf(Mrow[r], mx);
            alpha[r] = __expf(Mrow[r] - mnew);
            p0[r] = __expf(s0 - mnew);
            p1[r] = __expf(s1 - mnew);
            float rs = p0[r] + p1[r];
#pragma unroll
            for (int mk = 1; mk < 16; mk <<= 1) rs += __shfl_xor(rs, mk, 32);
            Lrow[r] = Lrow[r] * alpha[r] + rs;
            Mrow[r] = mnew;
        }
#pragma unroll
        for (int tt = 0; tt < 4; ++tt)
#pragma unroll
            for (int r = 0; r < 8; ++r) o4[tt][r] *= alpha[r];

        // P (16x32) -> per-wave LDS in bf16, then reload as A fragment
#pragma unroll
        for (int r = 0; r < 8; ++r) {
            int m = r + half * 8;
            sP[w][m * 32 + l16]      = f2bf(p0[r]);
            sP[w][m * 32 + 16 + l16] = f2bf(p1[r]);
        }
        asm volatile("s_wait_dscnt 0x0" ::: "memory");   // within-wave LDS turnaround
        bf16x16 ap = load_frag_a(&sP[w][l16 * 32], half);

        // O += P @ V  (4 d-tiles of 16)
#pragma unroll
        for (int tt = 0; tt < 4; ++tt) {
            bf16x16 bv = *(const bf16x16*)&sVt[(tt * 16 + l16) * 32 + half * 16];
            o4[tt] = wmma_bf16(ap, bv, o4[tt]);
        }
    }

    // Normalize and store bf16 attention output (head-major columns)
#pragma unroll
    for (int tt = 0; tt < 4; ++tt)
#pragma unroll
        for (int r = 0; r < 8; ++r) {
            int m = r + half * 8;
            float val = o4[tt][r] / Lrow[r];
            unsigned idx = (unsigned)(b * SEQ + qt * 16 + m) * DIM + h * DHEAD + tt * 16 + l16;
            O[idx] = f2bf(val);
        }
}

// ---------------------------------------------------------------------------
// Host launcher
// ---------------------------------------------------------------------------
extern "C" void kernel_launch(void* const* d_in, const int* in_sizes, int n_in,
                              void* d_out, int out_size, void* d_ws, size_t ws_size,
                              hipStream_t stream) {
    const float* x     = (const float*)d_in[0];
    const float* gamma = (const float*)d_in[1];
    const float* beta  = (const float*)d_in[2];
    const float* w_q   = (const float*)d_in[3];
    const float* w_k   = (const float*)d_in[4];
    const float* w_v   = (const float*)d_in[5];
    const float* w_o   = (const float*)d_in[6];
    const float* b_o   = (const float*)d_in[7];
    float* out = (float*)d_out;

    char* ws = (char*)d_ws;
    auto alloc = [&](size_t elems) {
        unsigned short* p = (unsigned short*)ws;
        ws += (elems * 2 + 255) & ~(size_t)255;
        return p;
    };
    unsigned short* xnb  = alloc((size_t)ROWS * DIM);     // normalized x, bf16
    unsigned short* wqbT = alloc((size_t)DIM * DIM);      // W_q^T [N][K]
    unsigned short* wkbT = alloc((size_t)DIM * KVDIM);
    unsigned short* wvbT = alloc((size_t)DIM * KVDIM);
    unsigned short* wobT = alloc((size_t)DIM * DIM);
    unsigned short* qb   = alloc((size_t)ROWS * DIM);
    unsigned short* kb   = alloc((size_t)ROWS * KVDIM);
    unsigned short* vb   = alloc((size_t)ROWS * KVDIM);
    unsigned short* ob   = alloc((size_t)ROWS * DIM);     // attention output, bf16

    // Weight cast + transpose (src [K][N] -> dst [N][K], bf16)
    castT_bf16_kernel<<<dim3(DIM / 32, DIM / 32),   256, 0, stream>>>(w_q, wqbT, DIM, DIM);
    castT_bf16_kernel<<<dim3(DIM / 32, KVDIM / 32), 256, 0, stream>>>(w_k, wkbT, DIM, KVDIM);
    castT_bf16_kernel<<<dim3(DIM / 32, KVDIM / 32), 256, 0, stream>>>(w_v, wvbT, DIM, KVDIM);
    castT_bf16_kernel<<<dim3(DIM / 32, DIM / 32),   256, 0, stream>>>(w_o, wobT, DIM, DIM);

    ln_bf16_kernel<<<ROWS, 256, 0, stream>>>(x, gamma, beta, xnb);

    // Projections (bf16 out)
    gemm_bf16_kernel<DIM, DIM, false><<<dim3(ROWS / 128, DIM / 64), 256, 0, stream>>>(
        xnb, wqbT, qb, nullptr, nullptr);
    gemm_bf16_kernel<KVDIM, DIM, false><<<dim3(ROWS / 128, KVDIM / 64), 256, 0, stream>>>(
        xnb, wkbT, kb, nullptr, nullptr);
    gemm_bf16_kernel<KVDIM, DIM, false><<<dim3(ROWS / 128, KVDIM / 64), 256, 0, stream>>>(
        xnb, wvbT, vb, nullptr, nullptr);

    // Attention: grid = B * KVHEADS * (SEQ/16) blocks of 4 waves
    attn_kernel<<<BATCH * KVHEADS * (SEQ / 16), 128, 0, stream>>>(qb, kb, vb, ob);

    // Output projection, fp32 + bias
    gemm_bf16_kernel<DIM, DIM, true><<<dim3(ROWS / 128, DIM / 64), 256, 0, stream>>>(
        ob, wobT, nullptr, out, b_o);
}
